// S4ModelHand_70506183131279
// MI455X (gfx1250) — matrix operand order repaired and checked by reading, not compiled, not run on hardware
//
#include <hip/hip_runtime.h>
#include <math.h>

// ---------------- problem constants ----------------
constexpr int cB    = 16;
constexpr int cL    = 2048;
constexpr int cDIN  = 128;
constexpr int cH    = 256;
constexpr int cN    = 64;
constexpr int cNL   = 4;
constexpr int cDOUT = 10;

typedef __bf16 bf16x16 __attribute__((ext_vector_type(16)));
typedef float  f32x8   __attribute__((ext_vector_type(8)));
typedef int    v4i     __attribute__((ext_vector_type(4)));
typedef unsigned short u16;

union FragAB { u16 h[16]; uint4 q[2]; bf16x16 v; };

__device__ __forceinline__ u16 f2bf(float f) {
  union { float f; unsigned u; } x; x.f = f;
  unsigned r = x.u + 0x7fffu + ((x.u >> 16) & 1u);   // round-to-nearest-even
  return (u16)(r >> 16);
}

// A-fragment K index for element j (16x32 bf16 A, ISA 7.12.2 layout)
//   elements 0..7 = K kh..kh+7, elements 8..15 = K kh+16..kh+23 (kh = 8*(lane>=16))
__device__ __forceinline__ int kmapA(int j, int lane) {
  int kh = (lane >> 4) << 3;
  return (j < 8) ? (kh + j) : (kh + j + 8);
}
// B-fragment K index (same as A: elements 0..7 = K kh..kh+7, 8..15 = K kh+16..kh+23)
__device__ __forceinline__ int kmapB(int j, int lane) {
  int kh = (lane >> 4) << 3;
  return (j < 8) ? (kh + j) : (kh + j + 8);
}

// Vectorized A-fragment load: base points at (row_start + kh) in a contiguous-K bf16 matrix.
__device__ __forceinline__ void loadA16(FragAB& a, const u16* base) {
  a.q[0] = *(const uint4*)(base);        // elements 0..7  : K kh..kh+7
  a.q[1] = *(const uint4*)(base + 16);   // elements 8..15 : K kh+16..kh+23
}
// Vectorized B-fragment load: base points at (col_start + kh) in a contiguous-K bf16 matrix.
__device__ __forceinline__ void loadB16(FragAB& b, const u16* base) {
  b.q[0] = *(const uint4*)(base);
  b.q[1] = *(const uint4*)(base + 16);
}

// ---- async global->LDS b128 copy (CDNA5), with synchronous fallback ----
typedef __attribute__((address_space(1))) v4i* gv4ip;   // global int4*
typedef __attribute__((address_space(3))) v4i* lv4ip;   // LDS int4*

__device__ __forceinline__ void async_copy16(const void* g, void* l) {
#if __has_builtin(__builtin_amdgcn_global_load_async_to_lds_b128)
  __builtin_amdgcn_global_load_async_to_lds_b128((gv4ip)(g), (lv4ip)(l), 0, 0);
#else
  *(uint4*)l = *(const uint4*)g;
#endif
}
__device__ __forceinline__ void async_wait0() {
#if __has_builtin(__builtin_amdgcn_s_wait_asynccnt)
  __builtin_amdgcn_s_wait_asynccnt(0);
#endif
}

// ---------------- f32 -> bf16 bulk convert ----------------
__global__ void cvt_bf16_kernel(const float* __restrict__ s, u16* __restrict__ d, int n) {
  int i = blockIdx.x * blockDim.x + threadIdx.x;
  if (i < n) d[i] = f2bf(s[i]);
}

// ---------------- enc_W transpose + convert: Wt[h*DIN+k] = enc_W[k*H+h] ----------------
__global__ void cvt_encWt_kernel(const float* __restrict__ W, u16* __restrict__ Wt) {
  int i = blockIdx.x * blockDim.x + threadIdx.x;   // H*DIN
  int h = i >> 7;
  int k = i & (cDIN - 1);
  Wt[i] = f2bf(W[k * cH + h]);
}

// ---------------- encoder: u[b,h,l] = x[b,l,:] @ enc_W + enc_b ----------------
__global__ __launch_bounds__(256) void enc_gemm_kernel(
    const u16* __restrict__ xbf, const u16* __restrict__ Wt,
    const float* __restrict__ bias, float* __restrict__ u, u16* __restrict__ ubf)
{
  const int lane = threadIdx.x & 31;
  const int tile = blockIdx.x * 8 + (threadIdx.x >> 5);   // 32768 tiles
  const int b  = tile >> 11;
  const int r  = tile & 2047;
  const int h0 = (r >> 7) << 4;
  const int l0 = (r & 127) << 4;
  const int m  = lane & 15;
  const int kh = (lane >> 4) << 3;
  const int kb16 = (lane >> 4) << 4;
  f32x8 c = {};
  for (int kk = 0; kk < cDIN; kk += 32) {
    FragAB a, bm;
    loadA16(a,  Wt  + (size_t)(h0 + m) * cDIN + kk + kh);                           // A[m,k] = enc_W[k,h0+m]
    loadB16(bm, xbf + (size_t)b * cL * cDIN + (size_t)(l0 + m) * cDIN + kk + kh); // B[k,n] = x[b,l0+n,k]
    c = __builtin_amdgcn_wmma_f32_16x16x32_bf16(false, a.v, false, bm.v, (short)0, c, false, false);
  }
  const int col = l0 + m;
#pragma unroll
  for (int i = 0; i < 8; ++i) {
    int row = h0 + i + ((lane >> 4) << 3);
    float v = c[i] + bias[row];
    size_t idx = (size_t)b * cH * cL + (size_t)row * cL + col;
    u[idx] = v; ubf[idx] = f2bf(v);
  }
}

// ---------------- per-layer S4D parameter prep ----------------
__global__ void prep_kernel(const float* __restrict__ log_dt, const float* __restrict__ log_A_real,
                            const float* __restrict__ A_imag, const float* __restrict__ C_re,
                            const float* __restrict__ C_im,
                            float* __restrict__ dre, float* __restrict__ dimv,
                            float* __restrict__ cbr, float* __restrict__ cbi)
{
  int i = blockIdx.x * blockDim.x + threadIdx.x;   // H*N
  int h = i >> 6;
  float dt = expf(log_dt[h]);
  float Ar = -expf(log_A_real[i]);
  float Ai = A_imag[i];
  float dr = Ar * dt, di = Ai * dt;
  float e  = expf(dr);
  float er = e * cosf(di) - 1.0f;                  // exp(dtA) - 1
  float ei = e * sinf(di);
  float mag = Ar * Ar + Ai * Ai;
  float tr = (er * Ar + ei * Ai) / mag;            // (exp(dtA)-1)/A
  float ti = (ei * Ar - er * Ai) / mag;
  float cr = C_re[i], ci = C_im[i];
  cbr[i] = cr * tr - ci * ti;                      // Cb = C * (exp(dtA)-1)/A
  cbi[i] = cr * ti + ci * tr;
  dre[i] = dr; dimv[i] = di;
}

// ---------------- Vandermonde: k[h,l] = 2*Re(sum_n Cb * exp(dtA*l)) ----------------
__global__ void vander_kernel(const float* __restrict__ dre, const float* __restrict__ dimv,
                              const float* __restrict__ cbr, const float* __restrict__ cbi,
                              u16* __restrict__ kk)
{
  int i = blockIdx.x * blockDim.x + threadIdx.x;   // H*L
  int h = i >> 11;
  int l = i & 2047;
  float fl = (float)l;
  float acc = 0.0f;
#pragma unroll 4
  for (int n = 0; n < cN; ++n) {
    float dr = dre[h * cN + n] * fl;
    float di = dimv[h * cN + n] * fl;
    float e  = expf(dr);
    float sn, cs; sincosf(di, &sn, &cs);
    acc += cbr[h * cN + n] * (e * cs) - cbi[h * cN + n] * (e * sn);
  }
  kk[i] = f2bf(2.0f * acc);
}

// ---------------- causal conv as Toeplitz GEMM ----------------
// y[b,h,l] = sum_{s<=l} k[h,l-s]*u[b,h,s].  M=batch(16), N=16 l's, K over s.
// All 8 waves of a block share head h -> stage k[h,:] (+32-elem zero pad) in LDS once.
__global__ __launch_bounds__(256) void conv_gemm_kernel(
    const u16* __restrict__ ubf, const u16* __restrict__ kker, float* __restrict__ y)
{
  __shared__ u16 kl[32 + cL];                      // zero pad kills the causal-mask branch
  const int tid  = threadIdx.x;
  const int hblk = (blockIdx.x * 8) >> 7;          // block-uniform head

  if (tid < 32) kl[tid] = 0;
  // 2048 halves = 4096B: each of 256 threads async-copies 16B
  async_copy16(kker + (size_t)hblk * cL + tid * 8, &kl[32 + tid * 8]);
  async_wait0();
  __syncthreads();

  const int lane = tid & 31;
  const int tile = blockIdx.x * 8 + (tid >> 5);    // H * L/16 = 32768 tiles
  const int l0 = (tile & 127) << 4;
  const int n  = lane & 15;                        // A-row = batch b ; B-col = l offset
  const int kh = (lane >> 4) << 3;
  f32x8 c = {};
  const int ksteps = (l0 + 16 + 31) >> 5;          // s stays < cL (l0 multiple of 16)
  for (int ks = 0; ks < ksteps; ++ks) {
    const int s0 = ks << 5;
    FragAB a, bm;
    loadA16(a, ubf + (size_t)n * cH * cL + (size_t)hblk * cL + s0 + kh);
#pragma unroll
    for (int j = 0; j < 16; ++j) {
      int d = l0 + n - (s0 + kmapB(j, lane));      // Toeplitz index; pad handles d<0
      bm.h[j] = kl[32 + d];
    }
    c = __builtin_amdgcn_wmma_f32_16x16x32_bf16(false, a.v, false, bm.v, (short)0, c, false, false);
  }
#pragma unroll
  for (int i = 0; i < 8; ++i) {
    int brow = i + ((lane >> 4) << 3);             // M dim = batch
    y[(size_t)brow * cH * cL + (size_t)hblk * cL + l0 + n] = c[i];
  }
}

// ---------------- elementwise: ybf = bf16(gelu(y + u*D_skip[h])) ----------------
__global__ void gelu_skip_kernel(const float* __restrict__ y, const float* __restrict__ u,
                                 const float* __restrict__ Dskip, u16* __restrict__ ybf)
{
  size_t i = (size_t)blockIdx.x * blockDim.x + threadIdx.x;  // B*H*L
  int h = (int)((i / cL) % cH);
  float v = y[i] + u[i] * Dskip[h];
  float g = 0.5f * v * (1.0f + erff(v * 0.70710678118654752f));
  ybf[i] = f2bf(g);
}

// ---------------- 1x1-conv + GLU + residual: z = a*sigmoid(g) + u ----------------
// Tile map: waves of a block share (b,l0), differ in o-tile -> B panel staged once in LDS.
__global__ __launch_bounds__(256) void outw_glu_kernel(
    const u16* __restrict__ Wbf, const float* __restrict__ bias,
    const u16* __restrict__ ybf, const float* __restrict__ u, float* __restrict__ z)
{
  __shared__ u16 yt[cH * 16];                      // 256 K-rows x 16 cols = 8KB
  const int tid   = threadIdx.x;
  const int tile0 = blockIdx.x * 8;
  const int b_blk = tile0 >> 11;
  const int l0blk = ((tile0 & 2047) >> 4) << 4;    // block-uniform l0

  // stage B panel: thread t copies row k=t (16 contiguous halves at [k*L + l0])
  {
    const u16* src = ybf + (size_t)b_blk * cH * cL + (size_t)tid * cL + l0blk;
    async_copy16(src,     &yt[tid * 16]);
    async_copy16(src + 8, &yt[tid * 16 + 8]);
  }
  async_wait0();
  __syncthreads();

  const int lane = tid & 31;
  const int tile = tile0 + (tid >> 5);
  const int o0 = (tile & 15) << 4;                 // o-tile varies across waves
  const int m  = lane & 15;
  const int kh = (lane >> 4) << 3;
  f32x8 ca = {}, cg = {};
  for (int kx = 0; kx < cH; kx += 32) {
    FragAB aa, ag, bm;
    loadA16(aa, Wbf + (size_t)(o0 + m) * cH + kx + kh);        // a-rows of out_W (bf16)
    loadA16(ag, Wbf + (size_t)(cH + o0 + m) * cH + kx + kh);   // g-rows
#pragma unroll
    for (int j = 0; j < 16; ++j)
      bm.h[j] = yt[(kx + kmapB(j, lane)) * 16 + m];
    ca = __builtin_amdgcn_wmma_f32_16x16x32_bf16(false, aa.v, false, bm.v, (short)0, ca, false, false);
    cg = __builtin_amdgcn_wmma_f32_16x16x32_bf16(false, ag.v, false, bm.v, (short)0, cg, false, false);
  }
  const int col = l0blk + m;
#pragma unroll
  for (int i = 0; i < 8; ++i) {
    int o = o0 + i + ((lane >> 4) << 3);
    float a = ca[i] + bias[o];
    float g = cg[i] + bias[cH + o];
    float sg = 1.0f / (1.0f + expf(-g));
    size_t idx = (size_t)b_blk * cH * cL + (size_t)o * cL + col;
    z[idx] = a * sg + u[idx];                      // GLU + residual
  }
}

// ---------------- channel LayerNorm: u = (z-m)/sqrt(v+eps)*w + b ----------------
__global__ void ln_kernel(const float* __restrict__ z, const float* __restrict__ w,
                          const float* __restrict__ bvec, float* __restrict__ u,
                          u16* __restrict__ ubf)
{
  int p = blockIdx.x * blockDim.x + threadIdx.x;   // B*L, coalesced over l
  int b = p >> 11;
  int l = p & 2047;
  size_t base = (size_t)b * cH * cL + l;
  float s = 0.0f, s2 = 0.0f;
  for (int h = 0; h < cH; ++h) {
    float v = z[base + (size_t)h * cL];
    s += v; s2 += v * v;
  }
  float mean = s * (1.0f / cH);
  float var  = s2 * (1.0f / cH) - mean * mean;
  float inv  = rsqrtf(var + 1e-5f);
  for (int h = 0; h < cH; ++h) {
    size_t idx = base + (size_t)h * cL;
    float v = (z[idx] - mean) * inv * w[h] + bvec[h];
    u[idx] = v; ubf[idx] = f2bf(v);
  }
}

// ---------------- decoder: out[b,l,o] = u[b,:,l] . dec_W[:,o] + dec_b ----------------
__global__ void dec_kernel(const float* __restrict__ u, const float* __restrict__ W,
                           const float* __restrict__ bias, float* __restrict__ out)
{
  int p = blockIdx.x * blockDim.x + threadIdx.x;   // B*L
  int b = p >> 11;
  int l = p & 2047;
  float acc[cDOUT];
#pragma unroll
  for (int o = 0; o < cDOUT; ++o) acc[o] = bias[o];
  size_t base = (size_t)b * cH * cL + l;
  for (int h = 0; h < cH; ++h) {
    float uv = u[base + (size_t)h * cL];
#pragma unroll
    for (int o = 0; o < cDOUT; ++o) acc[o] += uv * W[h * cDOUT + o];
  }
#pragma unroll
  for (int o = 0; o < cDOUT; ++o) out[(size_t)p * cDOUT + o] = acc[o];
}

// ---------------- host orchestration ----------------
extern "C" void kernel_launch(void* const* d_in, const int* in_sizes, int n_in,
                              void* d_out, int out_size, void* d_ws, size_t ws_size,
                              hipStream_t stream) {
  const float* x      = (const float*)d_in[0];
  const float* enc_W  = (const float*)d_in[1];
  const float* enc_b  = (const float*)d_in[2];
  const float* log_dt = (const float*)d_in[3];
  const float* logAr  = (const float*)d_in[4];
  const float* A_im   = (const float*)d_in[5];
  const float* C_re   = (const float*)d_in[6];
  const float* C_im   = (const float*)d_in[7];
  const float* Dsk    = (const float*)d_in[8];
  const float* out_W  = (const float*)d_in[9];
  const float* out_b  = (const float*)d_in[10];
  const float* ln_w   = (const float*)d_in[11];
  const float* ln_b   = (const float*)d_in[12];
  const float* dec_W  = (const float*)d_in[13];
  const float* dec_b  = (const float*)d_in[14];

  const size_t nBHL = (size_t)cB * cH * cL;        // 8.4M
  const int    nX   = cB * cL * cDIN;              // 4.2M
  const int    nOW  = cNL * 2 * cH * cH;           // 524288
  char* ws = (char*)d_ws;
  float* u_f32 = (float*)ws;  ws += nBHL * 4;
  u16*   u_bf  = (u16*)ws;    ws += nBHL * 2;
  float* y_f32 = (float*)ws;  ws += nBHL * 4;      // conv out, then reused as z (pre-LN)
  u16*   y_bf  = (u16*)ws;    ws += nBHL * 2;
  u16*   x_bf  = (u16*)ws;    ws += (size_t)nX * 2;
  u16*   ow_bf = (u16*)ws;    ws += (size_t)nOW * 2;
  u16*   Wt_bf = (u16*)ws;    ws += (size_t)cH * cDIN * 2;
  u16*   kker  = (u16*)ws;    ws += (size_t)cH * cL * 2;
  float* dre   = (float*)ws;  ws += (size_t)cH * cN * 4;
  float* dimv  = (float*)ws;  ws += (size_t)cH * cN * 4;
  float* cbr   = (float*)ws;  ws += (size_t)cH * cN * 4;
  float* cbi   = (float*)ws;  ws += (size_t)cH * cN * 4;

  // one-time bf16 operand prep (re-done every call; launch is stateless)
  cvt_bf16_kernel<<<nX / 256, 256, 0, stream>>>(x, x_bf, nX);
  cvt_bf16_kernel<<<nOW / 256, 256, 0, stream>>>(out_W, ow_bf, nOW);
  cvt_encWt_kernel<<<(cH * cDIN) / 256, 256, 0, stream>>>(enc_W, Wt_bf);

  // encoder
  enc_gemm_kernel<<<4096, 256, 0, stream>>>(x_bf, Wt_bf, enc_b, u_f32, u_bf);

  for (int layer = 0; layer < cNL; ++layer) {
    const size_t pHN = (size_t)layer * cH * cN;
    prep_kernel<<<(cH * cN) / 256, 256, 0, stream>>>(
        log_dt + (size_t)layer * cH, logAr + pHN, A_im + pHN, C_re + pHN, C_im + pHN,
        dre, dimv, cbr, cbi);
    vander_kernel<<<(cH * cL) / 256, 256, 0, stream>>>(dre, dimv, cbr, cbi, kker);
    conv_gemm_kernel<<<4096, 256, 0, stream>>>(u_bf, kker, y_f32);
    gelu_skip_kernel<<<(int)(nBHL / 256), 256, 0, stream>>>(y_f32, u_f32, Dsk + (size_t)layer * cH, y_bf);
    outw_glu_kernel<<<4096, 256, 0, stream>>>(
        ow_bf + (size_t)layer * 2 * cH * cH, out_b + (size_t)layer * 2 * cH,
        y_bf, u_f32, y_f32);
    ln_kernel<<<(cB * cL) / 256, 256, 0, stream>>>(
        y_f32, ln_w + (size_t)layer * cH, ln_b + (size_t)layer * cH, u_f32, u_bf);
  }

  dec_kernel<<<(cB * cL) / 256, 256, 0, stream>>>(u_f32, dec_W, dec_b, (float*)d_out);
}
